// LSTMCell_28913719836975
// MI455X (gfx1250) — matrix operand order repaired and checked by reading, not compiled, not run on hardware
//
#include <hip/hip_runtime.h>
#include <hip/hip_bf16.h>
#include <cstdint>
#include <cstddef>

typedef __attribute__((ext_vector_type(16))) __bf16 v16bf;
typedef __attribute__((ext_vector_type(8)))  float  v8f;

#define BROWS   131072
#define KDIM    512
#define HDIM    256
#define M_TILE  128          // batch rows per workgroup (8 waves x 16 rows)
#define COLS    128          // 4 gates x 32 h-columns staged per workgroup
#define COLSTR  260          // dword stride per staged row/col (1040 B: 16B aligned, 260%64==4)

#if __has_builtin(__builtin_amdgcn_sched_barrier)
#define SCHED_FENCE() __builtin_amdgcn_sched_barrier(0)
#else
#define SCHED_FENCE()
#endif

union BF16Frag {
    v16bf    v;
    uint32_t u[8];
    uint4    q[2];
};

// f32x2 -> packed bf16x2 (RTNE); lowers to v_cvt_pk_bf16_f32
__device__ __forceinline__ uint32_t pack_bf16x2(float a, float b) {
    union { __bf16 h[2]; uint32_t u; } p;
    p.h[0] = (__bf16)a;
    p.h[1] = (__bf16)b;
    return p.u;
}

__device__ __forceinline__ float fast_sigmoid(float x) {
    return 1.0f / (1.0f + __expf(-x));
}
__device__ __forceinline__ float fast_tanh(float x) {
    float e = __expf(-2.0f * x);
    return (1.0f - e) / (1.0f + e);
}

__global__ __launch_bounds__(256) void lstm_wmma_bf16_kernel(
    const float* __restrict__ x,  const float* __restrict__ h,  const float* __restrict__ c,
    const float* __restrict__ Wi, const float* __restrict__ bi,
    const float* __restrict__ Wf, const float* __restrict__ bf_,
    const float* __restrict__ Wo, const float* __restrict__ bo,
    const float* __restrict__ Wc, const float* __restrict__ bc,
    float* __restrict__ out, float* __restrict__ cnew)
{
    extern __shared__ uint32_t lds[];            // ldsW [COLS][COLSTR] ++ ldsA [M_TILE][COLSTR]
    uint32_t* ldsW = lds;
    uint32_t* ldsA = lds + COLS * COLSTR;

    const int tid   = threadIdx.x;
    const int wave  = tid >> 5;
    const int lane  = tid & 31;
    const int l16   = lane & 15;
    const int khalf = lane >> 4;                 // 0: k 0-7/16-23, 1: k 8-15/24-31 per k-step (ISA lane map)
    const int n0w   = blockIdx.y * 32;           // h-column block of this WG
    const int rowBlk  = blockIdx.x * M_TILE;
    const int rowBase = rowBlk + wave * 16;

    // ---------------- Stage W tile: 128 cols x K=512 -> packed bf16 in LDS -------------
    #pragma unroll 4
    for (int i = 0; i < 64; ++i) {
        int flat = i * 256 + tid;                // 16384 items
        int qk   = flat & 127;                   // float4 index along K
        int col  = flat >> 7;                    // 0..127
        int g    = col >> 5;
        int n    = n0w + (col & 31);
        const float* wrow = ((g == 0) ? Wi : (g == 1) ? Wf : (g == 2) ? Wo : Wc)
                            + (size_t)n * KDIM + 4 * qk;
        float4 w4 = *(const float4*)wrow;
        uint2 st;
        st.x = pack_bf16x2(w4.x, w4.y);
        st.y = pack_bf16x2(w4.z, w4.w);
        *(uint2*)&ldsW[col * COLSTR + 2 * qk] = st;
    }

    // ---------------- Stage A tile: 128 rows x (x||h) -> packed bf16 in LDS ------------
    #pragma unroll 4
    for (int i = 0; i < 64; ++i) {
        int flat = i * 256 + tid;
        int qk   = flat & 127;                   // 0..63 from x, 64..127 from h
        int row  = flat >> 7;                    // 0..127
        size_t grow = (size_t)(rowBlk + row) * 256;
        const float* src = (qk < 64) ? (x + grow + 4 * qk)
                                     : (h + grow + 4 * (qk - 64));
        float4 a4 = *(const float4*)src;
        uint2 st;
        st.x = pack_bf16x2(a4.x, a4.y);
        st.y = pack_bf16x2(a4.z, a4.w);
        *(uint2*)&ldsA[row * COLSTR + 2 * qk] = st;
    }
    __syncthreads();

    // ---------------- GEMM: 128 tile-ops, pinned software pipeline over LDS ------------
    v8f acc[4][2];
    {
        v8f z = {};
        #pragma unroll
        for (int g = 0; g < 4; ++g)
            #pragma unroll
            for (int j = 0; j < 2; ++j) acc[g][j] = z;
    }

    const uint32_t* aRow = &ldsA[(wave * 16 + l16) * COLSTR + khalf * 4];

    // Hoisted per-tile B base addresses: inner loop ds_loads use only immediates
    // (k-step offset = ks*64 (+32) bytes <= 1008, fits the 16-bit DS immediate).
    const uint32_t* bBase[8];
    #pragma unroll
    for (int t = 0; t < 8; ++t) {
        int col = (t >> 1) * 32 + (t & 1) * 16 + l16;
        bBase[t] = &ldsW[(size_t)col * COLSTR + khalf * 4];
    }

    #define LOAD_A(frag, kb) do {                                              \
        const uint4* _p = (const uint4*)(aRow + ((kb) >> 1));                  \
        (frag).q[0] = _p[0];  (frag).q[1] = _p[2];                             \
    } while (0)
    #define LOAD_B(frag, T) do {                                               \
        const uint4* _p = (const uint4*)(bBase[(T) & 7] + (((T) >> 3) * 16));  \
        (frag).q[0] = _p[0];  (frag).q[1] = _p[2];                             \
    } while (0)

    BF16Frag afr[2];
    BF16Frag bfr[4];                             // ring, prefetch distance 3
    LOAD_A(afr[0], 0);
    LOAD_B(bfr[0], 0);
    LOAD_B(bfr[1], 1);
    LOAD_B(bfr[2], 2);

    #pragma unroll
    for (int ks = 0; ks < 16; ++ks) {
        const int kb = ks * 32;
        if (ks < 15) LOAD_A(afr[(ks + 1) & 1], kb + 32);   // A for next k-step in flight
        #pragma unroll
        for (int t = 0; t < 8; ++t) {
            const int T = ks * 8 + t;
            if (T + 3 < 128) LOAD_B(bfr[(T + 3) & 3], T + 3);  // B three tiles ahead
            SCHED_FENCE();                                     // pin: loads stay above
            const int g = t >> 1, j = t & 1;
            acc[g][j] = __builtin_amdgcn_wmma_f32_16x16x32_bf16(
                false, afr[ks & 1].v, false, bfr[T & 3].v,
                (short)0, acc[g][j], false, false);
            SCHED_FENCE();                                     // pin: wmma stays here
        }
    }
    #undef LOAD_A
    #undef LOAD_B

    // ---------------- Fused LSTM epilogue ----------------------------------------------
    float bia[2], bfa[2], boa[2], bca[2];
    #pragma unroll
    for (int j = 0; j < 2; ++j) {
        int colH = n0w + j * 16 + l16;
        bia[j] = bi[colH];  bfa[j] = bf_[colH];
        boa[j] = bo[colH];  bca[j] = bc[colH];
    }

    #pragma unroll
    for (int j = 0; j < 2; ++j) {
        int colH = n0w + j * 16 + l16;
        #pragma unroll
        for (int r = 0; r < 8; ++r) {
            int m = r + khalf * 8;                    // C/D lane map: vgpr r -> row r(+8)
            size_t idx = (size_t)(rowBase + m) * HDIM + colH;
            float vi = acc[0][j][r] + bia[j];
            float vf = acc[1][j][r] + bfa[j];
            float vo = acc[2][j][r] + boa[j];
            float vg = acc[3][j][r] + bca[j];
            float si = fast_sigmoid(vi);
            float sf = fast_sigmoid(vf);
            float so = fast_sigmoid(vo);
            float tg = fast_tanh(vg);
            float cv = __builtin_nontemporal_load(c + idx);   // single-use: TH=NT
            float cn = sf * cv + si * tg;
            __builtin_nontemporal_store(fast_tanh(cn) * so, out + idx);   // streaming
            __builtin_nontemporal_store(cn, cnew + idx);
        }
    }
}

extern "C" void kernel_launch(void* const* d_in, const int* in_sizes, int n_in,
                              void* d_out, int out_size, void* d_ws, size_t ws_size,
                              hipStream_t stream) {
    const float* x  = (const float*)d_in[0];
    const float* h  = (const float*)d_in[1];
    const float* c  = (const float*)d_in[2];
    // d_in[3], d_in[4]: user_states / item_states (unused placeholders)
    const float* Wi = (const float*)d_in[5];
    const float* bi = (const float*)d_in[6];
    const float* Wf = (const float*)d_in[7];
    const float* bf = (const float*)d_in[8];
    const float* Wo = (const float*)d_in[9];
    const float* bo = (const float*)d_in[10];
    const float* Wc = (const float*)d_in[11];
    const float* bc = (const float*)d_in[12];

    float* out  = (float*)d_out;                       // tuple: (out, c_new) flat
    float* cnew = out + (size_t)BROWS * HDIM;

    dim3 grid(BROWS / M_TILE, HDIM / 32);              // 1024 x 8 workgroups
    dim3 block(256);                                   // 8 waves (wave32)
    size_t lds_bytes = (size_t)(COLS + M_TILE) * COLSTR * 4;   // 266,240 B (< 320 KB/WGP)

    lstm_wmma_bf16_kernel<<<grid, block, lds_bytes, stream>>>(
        x, h, c, Wi, bi, Wf, bf, Wo, bo, Wc, bc, out, cnew);
}
